// Head_81346680586561
// MI455X (gfx1250) — matrix operand order repaired and checked by reading, not compiled, not run on hardware
//
#include <hip/hip_runtime.h>
#include <hip/hip_bf16.h>
#include <math.h>

// MI455X / gfx1250 fused causal attention.
// B=8, T=2048, E=1024, H=128. bf16 WMMA (16x16x32) with f32 accumulation.
// Every WMMA fragment is a single contiguous 16B/32B vector load; fragment
// loads are batched ahead of WMMA bursts so one s_wait covers a whole clause.

typedef __bf16 bf16_t;
typedef __attribute__((ext_vector_type(16))) __bf16 v16bf;
typedef __attribute__((ext_vector_type(8)))  __bf16 v8bf;
typedef __attribute__((ext_vector_type(8)))  float  v8f;

#define B_SZ 8
#define T_SZ 2048
#define E_SZ 1024
#define H_SZ 128
#define ROWS (B_SZ * T_SZ)   // 16384

static __device__ __forceinline__ v8f wmma_bf16(v16bf a, v16bf b, v8f c) {
  // D = A(16x32) * B(32x16) + C(16x16), f32 accumulate
  return __builtin_amdgcn_wmma_f32_16x16x32_bf16(
      /*neg_a=*/false, a, /*neg_b=*/false, b,
      /*c_mod=*/(short)0, c, /*reuse_a=*/false, /*reuse_b=*/false);
}

// ---------------------------------------------------------------------------
// Kernel 0: pack W (f32 [E,H] x3) into bf16 transposed Wt[mat][h][e].
// 768 KB total -> permanently L2-resident (192 MB L2); every projection
// B-fragment becomes one contiguous 32-byte load.
// ---------------------------------------------------------------------------
__global__ __launch_bounds__(256) void pack_w_kernel(
    const float* __restrict__ Wq, const float* __restrict__ Wk,
    const float* __restrict__ Wv, bf16_t* __restrict__ wt) {
  const int idx = blockIdx.x * 256 + threadIdx.x;   // 0 .. 3*H*E-1
  const int mat = idx / (H_SZ * E_SZ);
  const int rem = idx - mat * (H_SZ * E_SZ);
  const int h   = rem / E_SZ;
  const int e   = rem - h * E_SZ;
  const float* W = (mat == 0) ? Wq : (mat == 1) ? Wk : Wv;
  wt[idx] = (bf16_t)W[(size_t)e * H_SZ + h];
}

// ---------------------------------------------------------------------------
// Kernel 1: Q/K/V projection.  One wave computes a [16 x 128] output tile for
// one of {Wq,Wk,Wv}.  Per k-step: load A + all 8 B fragments (one clause),
// then 8 back-to-back WMMAs.  Q,K stored [b*T+t][h] bf16; V stored transposed
// [b][h][t] bf16 so the attention P*V B-fragments are contiguous.
// ---------------------------------------------------------------------------
__global__ __launch_bounds__(256) void qkv_proj_kernel(
    const float* __restrict__ x,
    const bf16_t* __restrict__ wt,     // [3][H][E] bf16
    bf16_t* __restrict__ qkv) {
  const int lane    = threadIdx.x & 31;
  const int wid     = threadIdx.x >> 5;
  const int rowTile = blockIdx.x * 8 + wid;       // 0..1023
  const int mat     = blockIdx.y;                 // 0=q,1=k,2=v
  const bf16_t* wtm = wt + (size_t)mat * H_SZ * E_SZ;

  const int laneHi = lane >> 4;   // half-wave select
  const int laneLo = lane & 15;

  v8f acc[8];
  #pragma unroll
  for (int i = 0; i < 8; ++i)
    #pragma unroll
    for (int j = 0; j < 8; ++j) acc[i][j] = 0.0f;

  // A layout: row = lane%16; lanes 0-15 carry K {0..15},
  // lanes 16-31 carry K {16..31}
  const int arow = rowTile * 16 + laneLo;
  const float* xrow = x + (size_t)arow * E_SZ;

  for (int kt = 0; kt < E_SZ / 32; ++kt) {
    // ---- batch all fragment loads, then one WMMA burst ----
    const int kb = kt * 32 + laneHi * 16;
    const int kkb = kt * 32 + laneHi * 16;

    v16bf bf[8];
    #pragma unroll
    for (int nt = 0; nt < 8; ++nt)
      bf[nt] = *(const v16bf*)(wtm + (size_t)(nt * 16 + laneLo) * E_SZ + kkb);

    v16bf a;
    #pragma unroll
    for (int e = 0; e < 8; ++e) {
      a[e]     = (bf16_t)xrow[kb + e];        // contiguous 8 f32
      a[e + 8] = (bf16_t)xrow[kb + 8 + e];    // contiguous 8 f32
    }

    #pragma unroll
    for (int nt = 0; nt < 8; ++nt)
      acc[nt] = wmma_bf16(a, bf[nt], acc[nt]);
  }

  // C layout: VGPR r holds M = r (lanes 0-15) / r+8 (lanes 16-31), N = lane%16
  const int mrow = rowTile * 16 + laneHi * 8;
  if (mat < 2) {
    bf16_t* outBase = qkv + (size_t)mat * ROWS * H_SZ;
    #pragma unroll
    for (int nt = 0; nt < 8; ++nt)
      #pragma unroll
      for (int r = 0; r < 8; ++r)
        outBase[(size_t)(mrow + r) * H_SZ + nt * 16 + laneLo] = (bf16_t)acc[nt][r];
  } else {
    // V transposed: vt[b][h][t], rows r contiguous along t -> packed 16B store
    bf16_t* vt = qkv + 2 * (size_t)ROWS * H_SZ;
    const int b = (rowTile * 16) / T_SZ;
    const int t0 = mrow - b * T_SZ;
    #pragma unroll
    for (int nt = 0; nt < 8; ++nt) {
      const int col = nt * 16 + laneLo;
      v8bf pk;
      #pragma unroll
      for (int r = 0; r < 8; ++r) pk[r] = (bf16_t)acc[nt][r];
      *(v8bf*)(vt + ((size_t)b * H_SZ + col) * T_SZ + t0) = pk;
    }
  }
}

// ---------------------------------------------------------------------------
// Kernel 2: flash attention.  One wave owns a 16-query tile; iterates over
// 32-key chunks up to the causal diagonal.  S = Q*K^T (8 WMMA / chunk),
// log2-domain online softmax, O += P*V (8 WMMA / chunk).  P converted
// C-layout -> A-layout through per-wave LDS staging.
// Heaviest q-tiles are mapped to the earliest waves (reverse qt order).
// ---------------------------------------------------------------------------
__global__ __launch_bounds__(256) void attn_kernel(
    const bf16_t* __restrict__ qkv, float* __restrict__ out) {
  __shared__ bf16_t Plds[8][16][32];   // per-wave 16x32 P tile (8 KB)

  const int lane   = threadIdx.x & 31;
  const int wid    = threadIdx.x >> 5;
  const int w      = blockIdx.x * 8 + wid;   // 0..1023
  const int b      = w >> 7;                 // batch
  const int qt     = 127 - (w & 127);        // heavy tiles launch first
  const int laneHi = lane >> 4;
  const int laneLo = lane & 15;

  const bf16_t* q_ws = qkv;
  const bf16_t* k_ws = qkv + (size_t)ROWS * H_SZ;
  const bf16_t* vt   = qkv + 2 * (size_t)ROWS * H_SZ;   // [b][h][t]
  const size_t  brow = (size_t)b * T_SZ;

  // Preload Q A-fragments for the full head dim (4 x K=32)
  const int qrow = qt * 16 + laneLo;
  const bf16_t* qrp = q_ws + (brow + qrow) * H_SZ;
  v16bf qf[4];
  #pragma unroll
  for (int ks = 0; ks < 4; ++ks) {
    const int kb = ks * 32 + laneHi * 16;
    const v8bf lo = *(const v8bf*)(qrp + kb);
    const v8bf hi = *(const v8bf*)(qrp + kb + 8);
    #pragma unroll
    for (int e = 0; e < 8; ++e) { qf[ks][e] = lo[e]; qf[ks][e + 8] = hi[e]; }
  }

  v8f o[8];
  float mrow[8], lrow[8];
  #pragma unroll
  for (int i = 0; i < 8; ++i) {
    #pragma unroll
    for (int j = 0; j < 8; ++j) o[i][j] = 0.0f;
    mrow[i] = -__builtin_inff();
    lrow[i] = 0.0f;
  }

  // log2-domain softmax: fold 1/sqrt(H) and log2(e) into one scale
  const float scale2 = 0.08838834764831845f * 1.4426950408889634f;
  const int nchunks = (qt * 16 + 47) >> 5;   // cover keys 0..qt*16+15

  for (int c = 0; c < nchunks; ++c) {
    const int keybase = c * 32;

    // Prefetch next chunk's K rows into the cache hierarchy
    if (c + 1 < nchunks)
      __builtin_prefetch(k_ws + (brow + keybase + 32 + laneLo) * H_SZ, 0, 1);

    // ---- batch all 8 K B-fragments (contiguous 32B each), then WMMA burst ----
    v16bf kf[2][4];
    #pragma unroll
    for (int t = 0; t < 2; ++t) {
      const bf16_t* krp = k_ws + (brow + keybase + t * 16 + laneLo) * H_SZ;
      #pragma unroll
      for (int ks = 0; ks < 4; ++ks)
        kf[t][ks] = *(const v16bf*)(krp + ks * 32 + laneHi * 16);
    }

    v8f S[2];
    #pragma unroll
    for (int t = 0; t < 2; ++t) {
      v8f s;
      #pragma unroll
      for (int j = 0; j < 8; ++j) s[j] = 0.0f;
      #pragma unroll
      for (int ks = 0; ks < 4; ++ks)
        s = wmma_bf16(qf[ks], kf[t][ks], s);
      S[t] = s;
    }

    // ---- scale (log2 domain) + causal mask ----
    #pragma unroll
    for (int t = 0; t < 2; ++t)
      #pragma unroll
      for (int r = 0; r < 8; ++r) {
        const int row = qt * 16 + r + laneHi * 8;
        const int col = keybase + t * 16 + laneLo;
        const float v = S[t][r] * scale2;
        S[t][r] = (col <= row) ? v : -__builtin_inff();
      }

    // ---- online softmax (row stats per VGPR slot, reduce over 16 lanes) ----
    float pscale[8];
    #pragma unroll
    for (int r = 0; r < 8; ++r) {
      float vmax = fmaxf(S[0][r], S[1][r]);
      #pragma unroll
      for (int m = 8; m >= 1; m >>= 1)
        vmax = fmaxf(vmax, __shfl_xor(vmax, m, 32));   // stays in 16-lane half
      const float mnew = fmaxf(mrow[r], vmax);
      const float sc   = exp2f(mrow[r] - mnew);        // bare v_exp_f32
      mrow[r] = mnew;
      const float p0 = exp2f(S[0][r] - mnew);
      const float p1 = exp2f(S[1][r] - mnew);
      S[0][r] = p0; S[1][r] = p1;
      float rsum = p0 + p1;
      #pragma unroll
      for (int m = 8; m >= 1; m >>= 1)
        rsum += __shfl_xor(rsum, m, 32);
      lrow[r] = lrow[r] * sc + rsum;
      pscale[r] = sc;
    }
    #pragma unroll
    for (int nt = 0; nt < 8; ++nt)
      #pragma unroll
      for (int r = 0; r < 8; ++r) o[nt][r] *= pscale[r];

    // ---- P: C-layout -> A-layout via per-wave LDS staging ----
    #pragma unroll
    for (int t = 0; t < 2; ++t)
      #pragma unroll
      for (int r = 0; r < 8; ++r)
        Plds[wid][r + laneHi * 8][t * 16 + laneLo] = (bf16_t)S[t][r];
    asm volatile("s_wait_dscnt 0" ::: "memory");  // DS in-order per wave; order VALU reads

    v16bf pf;
    {
      const bf16_t* pr = &Plds[wid][laneLo][0];
      const int kb = laneHi * 16;
      #pragma unroll
      for (int e = 0; e < 8; ++e) {
        pf[e]     = pr[kb + e];
        pf[e + 8] = pr[kb + 8 + e];
      }
    }

    // ---- batch all 8 V B-fragments (Vt[b][h][t], contiguous 32B), then burst ----
    v16bf vf[8];
    #pragma unroll
    for (int nt = 0; nt < 8; ++nt)
      vf[nt] = *(const v16bf*)(
          vt + ((size_t)b * H_SZ + nt * 16 + laneLo) * T_SZ + keybase + laneHi * 16);
    #pragma unroll
    for (int nt = 0; nt < 8; ++nt)
      o[nt] = wmma_bf16(pf, vf[nt], o[nt]);
  }

  // ---- epilogue: normalize (8 reciprocals, then multiplies) and store ----
  float linv[8];
  #pragma unroll
  for (int r = 0; r < 8; ++r) linv[r] = 1.0f / lrow[r];
  #pragma unroll
  for (int nt = 0; nt < 8; ++nt)
    #pragma unroll
    for (int r = 0; r < 8; ++r) {
      const int row = qt * 16 + r + laneHi * 8;
      out[(brow + row) * H_SZ + nt * 16 + laneLo] = o[nt][r] * linv[r];
    }
}

// ---------------------------------------------------------------------------
extern "C" void kernel_launch(void* const* d_in, const int* in_sizes, int n_in,
                              void* d_out, int out_size, void* d_ws, size_t ws_size,
                              hipStream_t stream) {
  const float* x  = (const float*)d_in[0];
  const float* Wq = (const float*)d_in[1];
  const float* Wk = (const float*)d_in[2];
  const float* Wv = (const float*)d_in[3];
  float* out = (float*)d_out;

  // ws layout (bf16): [q: ROWS*H][k: ROWS*H][vt: ROWS*H][wt: 3*H*E]
  bf16_t* qkv = (bf16_t*)d_ws;                       // 12 MB
  bf16_t* wt  = qkv + 3 * (size_t)ROWS * H_SZ;       // +0.75 MB

  (void)in_sizes; (void)n_in; (void)out_size; (void)ws_size;

  // Pack weights: 3*H*E elements, 256/block
  pack_w_kernel<<<dim3((3 * H_SZ * E_SZ) / 256), 256, 0, stream>>>(Wq, Wk, Wv, wt);
  // 1024 row-tiles, 8 waves/block -> 128 blocks; y = matrix select
  qkv_proj_kernel<<<dim3(128, 3), 256, 0, stream>>>(x, wt, qkv);
  // 1024 (b, qtile) waves, 8 waves/block -> 128 blocks
  attn_kernel<<<dim3(128), 256, 0, stream>>>(qkv, out);
}
